// HGTLayer_4286377361444
// MI455X (gfx1250) — matrix operand order repaired and checked by reading, not compiled, not run on hardware
//
#include <hip/hip_runtime.h>
#include <hip/hip_bf16.h>
#include <stdint.h>

#define HID 256
#define NHEAD 8
#define HDIM 32
#define LN_EPS 1e-5f

typedef __attribute__((ext_vector_type(16))) __bf16          v16bf;
typedef __attribute__((ext_vector_type(8)))  float           v8f;
typedef __attribute__((ext_vector_type(16))) unsigned short  v16u;
typedef __attribute__((ext_vector_type(8)))  unsigned short  u16x8;

static __device__ __forceinline__ unsigned short f2bf(float f) {
    unsigned int u = __float_as_uint(f);
    u += 0x7fffu + ((u >> 16) & 1u);          // round-to-nearest-even
    return (unsigned short)(u >> 16);
}
static __device__ __forceinline__ float bf2f(unsigned short h) {
    return __uint_as_float(((unsigned int)h) << 16);
}
static __device__ __forceinline__ void atomicMaxF(float* addr, float val) {
    int old = __float_as_int(*addr);
    while (__int_as_float(old) < val) {
        int assumed = old;
        old = atomicCAS((int*)addr, assumed, __float_as_int(val));
        if (old == assumed) break;
    }
}

// ---------------------------------------------------------------------------
// Pack a 256x256 weight (optionally folding the per-head relation transform)
// into bf16 WMMA B-fragment order:
//   frag (nt,kt): lane l holds column n = nt*16 + (l&15),
//                 elems e=0..15 -> k = kt*32 + (l>>4)*16 + e
// so the GEMM does one contiguous 32B load per lane per fragment.
// ---------------------------------------------------------------------------
__global__ __launch_bounds__(128) void pack_weight(
    const float* __restrict__ W, const float* __restrict__ rel, int has_rel,
    unsigned short* __restrict__ out)
{
    const int lane = threadIdx.x & 31;
    const int frag = blockIdx.x * 4 + (threadIdx.x >> 5);   // 0..127
    const int nt = frag >> 3, kt = frag & 7;
    const int n  = nt * 16 + (lane & 15);
    const int h  = n >> 5, eo = n & 31;
    const int kb = kt * 32 + ((lane >> 4) << 4);
    unsigned short* o = out + ((size_t)frag * 32 + lane) * 16;
    #pragma unroll
    for (int e = 0; e < 16; ++e) {
        const int k = kb + e;
        float v;
        if (has_rel) {
            // W_eff[k][h*32+eo] = sum_d W[k][h*32+d] * rel[h][d][eo]
            const float* wr = W + (size_t)k * HID + h * HDIM;
            const float* rr = rel + (size_t)h * HDIM * HDIM + eo;
            float acc = 0.f;
            #pragma unroll 4
            for (int d = 0; d < HDIM; ++d) acc = fmaf(wr[d], rr[(size_t)d * HDIM], acc);
            v = acc;
        } else {
            v = W[(size_t)k * HID + n];
        }
        o[e] = f2bf(v);
    }
}

// ---------------------------------------------------------------------------
// Fused QKV projection: [rows x 256] x three 256x256 weights -> bf16 outputs.
// Block = 512 threads = 16 waves; wave nt owns output column tile nt.
// M-register-blocked: 64 rows per block = 4 M-tiles per wave, so each B
// fragment (global b128 x2) is reused by 4 WMMAs -> 4x less L2 B-traffic.
// A tile (64 rows x 256 K) staged in LDS as bf16, row pitch 264 for bank
// rotation under ds_load_b128. Per k-chunk: 6 global b128 + 8 ds b128 loads
// feed 12 v_wmma_f32_16x16x32_bf16, accumulating in f32.
// ---------------------------------------------------------------------------
__global__ __launch_bounds__(512) void gemm_qkv(
    const float* __restrict__ X,
    const unsigned short* __restrict__ Wq, const unsigned short* __restrict__ Wk,
    const unsigned short* __restrict__ Wv,
    unsigned short* __restrict__ Qo, unsigned short* __restrict__ Ko,
    unsigned short* __restrict__ Vo, int n_rows)
{
    __shared__ unsigned short As[64 * 264];
    const int tid  = threadIdx.x;
    const int lane = tid & 31;
    const int nt   = tid >> 5;              // wave == column tile 0..15
    const int row0 = blockIdx.x * 64;

    #pragma unroll
    for (int it = 0; it < 32; ++it) {
        const int i = tid + it * 512;       // 0..16383, coalesced f32 loads
        const int r = i >> 8, c = i & 255;
        const float v = (row0 + r < n_rows) ? X[(size_t)(row0 + r) * HID + c] : 0.f;
        As[r * 264 + c] = f2bf(v);
    }
    __syncthreads();

    v8f accQ[4] = {}, accK[4] = {}, accV[4] = {};
    const int r    = lane & 15;
    const int koff = (lane >> 4) << 4;      // contiguous K half-offset (matches packed B)
    #pragma unroll
    for (int kt = 0; kt < 8; ++kt) {
        // B fragments for this k-chunk: loaded once, reused by 4 M-tiles.
        const size_t boff = ((size_t)(nt * 8 + kt) * 32 + lane) * 16;
        const v16bf Bq = __builtin_bit_cast(v16bf, *(const v16u*)(Wq + boff));
        const v16bf Bk = __builtin_bit_cast(v16bf, *(const v16u*)(Wk + boff));
        const v16bf Bv = __builtin_bit_cast(v16bf, *(const v16u*)(Wv + boff));
        #pragma unroll
        for (int mt = 0; mt < 4; ++mt) {
            // A fragment: elems 0..15 -> K=koff+0..15
            const int base = (mt * 16 + r) * 264 + kt * 32 + koff;
            const u16x8 alo = *(const u16x8*)(&As[base]);
            const u16x8 ahi = *(const u16x8*)(&As[base + 8]);
            v16u ab;
            #pragma unroll
            for (int j = 0; j < 8; ++j) { ab[j] = alo[j]; ab[8 + j] = ahi[j]; }
            const v16bf A = __builtin_bit_cast(v16bf, ab);
            accQ[mt] = __builtin_amdgcn_wmma_f32_16x16x32_bf16(false, A, false, Bq, (short)0, accQ[mt], false, false);
            accK[mt] = __builtin_amdgcn_wmma_f32_16x16x32_bf16(false, A, false, Bk, (short)0, accK[mt], false, false);
            accV[mt] = __builtin_amdgcn_wmma_f32_16x16x32_bf16(false, A, false, Bv, (short)0, accV[mt], false, false);
        }
    }

    const int col = nt * 16 + (lane & 15);
    #pragma unroll
    for (int mt = 0; mt < 4; ++mt) {
        const int rbase = row0 + mt * 16 + ((lane >> 4) << 3);
        #pragma unroll
        for (int f = 0; f < 8; ++f) {
            const int rr = rbase + f;
            if (rr < n_rows) {
                const size_t o = (size_t)rr * HID + col;
                Qo[o] = f2bf(accQ[mt][f]);
                Ko[o] = f2bf(accK[mt][f]);
                Vo[o] = f2bf(accV[mt][f]);
            }
        }
    }
}

// ---------------------------------------------------------------------------
__global__ void init_buffers(float* m_ui, float* m_iu, float* den_ui, float* den_iu,
                             float* agg_item, float* agg_user, int n)
{
    const int i = blockIdx.x * blockDim.x + threadIdx.x;
    if (i < n * HID) { agg_item[i] = 0.f; agg_user[i] = 0.f; }
    if (i < n * NHEAD) {
        const float ninf = __uint_as_float(0xff800000u);
        m_ui[i] = ninf; m_iu[i] = ninf;
        den_ui[i] = 0.f; den_iu[i] = 0.f;
    }
}

// ---------------------------------------------------------------------------
// Pass 1: per-edge scores + segment max. One edge per wave32:
// lane covers channels [8l,8l+8), head = lane>>2; 4-lane shfl_xor dot-reduce.
// ---------------------------------------------------------------------------
__global__ __launch_bounds__(256) void edge_score(
    const unsigned short* __restrict__ Kp, const unsigned short* __restrict__ Q,
    const int* __restrict__ src, const int* __restrict__ dst,
    const float* __restrict__ mu, float* __restrict__ s, float* __restrict__ m, int E)
{
    const int e = blockIdx.x * 8 + (threadIdx.x >> 5);
    if (e >= E) return;
    const int lane = threadIdx.x & 31;
    __builtin_prefetch(src + e + 2048, 0, 0);
    __builtin_prefetch(dst + e + 2048, 0, 0);
    const int si = src[e], di = dst[e];
    const u16x8 kv = *(const u16x8*)(Kp + (size_t)si * HID + lane * 8);
    const u16x8 qv = *(const u16x8*)(Q  + (size_t)di * HID + lane * 8);
    float acc = 0.f;
    #pragma unroll
    for (int j = 0; j < 8; ++j) acc = fmaf(bf2f(kv[j]), bf2f(qv[j]), acc);
    acc += __shfl_xor(acc, 1, 32);
    acc += __shfl_xor(acc, 2, 32);
    if ((lane & 3) == 0) {
        const int h = lane >> 2;
        const float sc = acc * mu[h] * 0.17677669529663687f;   // 1/sqrt(32)
        s[(size_t)e * NHEAD + h] = sc;
        atomicMaxF(m + (size_t)di * NHEAD + h, sc);
    }
}

// ---------------------------------------------------------------------------
// Pass 2: unnormalized exp weights -> denominator + weighted value scatter.
// ---------------------------------------------------------------------------
__global__ __launch_bounds__(256) void edge_accum(
    const unsigned short* __restrict__ Vp,
    const float* __restrict__ s, const float* __restrict__ m,
    const int* __restrict__ src, const int* __restrict__ dst,
    float* __restrict__ den, float* __restrict__ agg, int E)
{
    const int e = blockIdx.x * 8 + (threadIdx.x >> 5);
    if (e >= E) return;
    const int lane = threadIdx.x & 31;
    const int si = src[e], di = dst[e];
    const int h = lane >> 2;
    const float w = __expf(s[(size_t)e * NHEAD + h] - m[(size_t)di * NHEAD + h]);
    if ((lane & 3) == 0) atomicAdd(den + (size_t)di * NHEAD + h, w);
    const u16x8 vv = *(const u16x8*)(Vp + (size_t)si * HID + lane * 8);
    float* ap = agg + (size_t)di * HID + lane * 8;
    #pragma unroll
    for (int j = 0; j < 8; ++j) atomicAdd(ap + j, w * bf2f(vv[j]));
}

// ---------------------------------------------------------------------------
// Softmax normalize (fused) + residual + LayerNorm + ELU. One node per wave.
// ---------------------------------------------------------------------------
__global__ __launch_bounds__(256) void out_block(
    const float* __restrict__ agg, const float* __restrict__ den,
    const float* __restrict__ hres, const float* __restrict__ g,
    const float* __restrict__ b, float* __restrict__ out, int n_rows)
{
    const int n = blockIdx.x * 8 + (threadIdx.x >> 5);
    if (n >= n_rows) return;
    const int lane = threadIdx.x & 31;
    const int h = lane >> 2;
    const float d = den[(size_t)n * NHEAD + h];
    const float inv = (d > 0.f) ? 1.f / d : 0.f;
    const float* ar = agg  + (size_t)n * HID + lane * 8;
    const float* hr = hres + (size_t)n * HID + lane * 8;
    float y[8];
    float lsum = 0.f;
    #pragma unroll
    for (int j = 0; j < 8; ++j) { y[j] = ar[j] * inv + hr[j]; lsum += y[j]; }
    #pragma unroll
    for (int o = 16; o > 0; o >>= 1) lsum += __shfl_xor(lsum, o, 32);
    const float mean = lsum * (1.f / HID);
    float lv = 0.f;
    #pragma unroll
    for (int j = 0; j < 8; ++j) { const float t = y[j] - mean; lv += t * t; }
    #pragma unroll
    for (int o = 16; o > 0; o >>= 1) lv += __shfl_xor(lv, o, 32);
    const float rstd = rsqrtf(lv * (1.f / HID) + LN_EPS);
    #pragma unroll
    for (int j = 0; j < 8; ++j) {
        const int c = lane * 8 + j;
        const float v = (y[j] - mean) * rstd * g[c] + b[c];
        out[(size_t)n * HID + c] = (v > 0.f) ? v : (__expf(v) - 1.f);
    }
}

// ---------------------------------------------------------------------------
extern "C" void kernel_launch(void* const* d_in, const int* in_sizes, int n_in,
                              void* d_out, int out_size, void* d_ws, size_t ws_size,
                              hipStream_t stream)
{
    const float* h_user   = (const float*)d_in[0];
    const float* h_item   = (const float*)d_in[1];
    const float* Wq_u     = (const float*)d_in[2];
    const float* Wk_u     = (const float*)d_in[3];
    const float* Wv_u     = (const float*)d_in[4];
    const float* Wq_i     = (const float*)d_in[5];
    const float* Wk_i     = (const float*)d_in[6];
    const float* Wv_i     = (const float*)d_in[7];
    const float* relWk_ui = (const float*)d_in[8];
    const float* relWv_ui = (const float*)d_in[9];
    const float* mu_ui    = (const float*)d_in[10];
    const float* relWk_iu = (const float*)d_in[11];
    const float* relWv_iu = (const float*)d_in[12];
    const float* mu_iu    = (const float*)d_in[13];
    const float* ln_g_u   = (const float*)d_in[14];
    const float* ln_b_u   = (const float*)d_in[15];
    const float* ln_g_i   = (const float*)d_in[16];
    const float* ln_b_i   = (const float*)d_in[17];
    const int* src_ui     = (const int*)d_in[18];
    const int* dst_ui     = (const int*)d_in[19];
    const int* src_iu     = (const int*)d_in[20];
    const int* dst_iu     = (const int*)d_in[21];

    const int N    = in_sizes[0] / HID;
    const int E_ui = in_sizes[18];
    const int E_iu = in_sizes[20];

    char* wp = (char*)d_ws;
    auto take = [&](size_t bytes) {
        char* p = wp; wp += (bytes + 255) & ~(size_t)255; return p;
    };
    unsigned short* pw[6];
    for (int i = 0; i < 6; ++i) pw[i] = (unsigned short*)take((size_t)65536 * 2);
    const size_t nodeBF = (size_t)N * HID * sizeof(unsigned short);
    unsigned short* Qu  = (unsigned short*)take(nodeBF);
    unsigned short* Kpu = (unsigned short*)take(nodeBF);
    unsigned short* Vpu = (unsigned short*)take(nodeBF);
    unsigned short* Qi  = (unsigned short*)take(nodeBF);
    unsigned short* Kpi = (unsigned short*)take(nodeBF);
    unsigned short* Vpi = (unsigned short*)take(nodeBF);
    float* s_ui   = (float*)take((size_t)E_ui * NHEAD * 4);
    float* s_iu   = (float*)take((size_t)E_iu * NHEAD * 4);
    float* m_ui   = (float*)take((size_t)N * NHEAD * 4);
    float* m_iu   = (float*)take((size_t)N * NHEAD * 4);
    float* den_ui = (float*)take((size_t)N * NHEAD * 4);
    float* den_iu = (float*)take((size_t)N * NHEAD * 4);
    float* agg_item = (float*)take((size_t)N * HID * 4);
    float* agg_user = (float*)take((size_t)N * HID * 4);

    // --- pack weights (relation transforms folded into K/V weights) ---
    pack_weight<<<32, 128, 0, stream>>>(Wq_u, nullptr,  0, pw[0]);
    pack_weight<<<32, 128, 0, stream>>>(Wk_u, relWk_ui, 1, pw[1]);
    pack_weight<<<32, 128, 0, stream>>>(Wv_u, relWv_ui, 1, pw[2]);
    pack_weight<<<32, 128, 0, stream>>>(Wq_i, nullptr,  0, pw[3]);
    pack_weight<<<32, 128, 0, stream>>>(Wk_i, relWk_iu, 1, pw[4]);
    pack_weight<<<32, 128, 0, stream>>>(Wv_i, relWv_iu, 1, pw[5]);

    // --- fused projections (WMMA bf16, f32 accumulate, 64-row M-blocking) ---
    const int rowBlocks = (N + 63) / 64;
    gemm_qkv<<<rowBlocks, 512, 0, stream>>>(h_user, pw[0], pw[1], pw[2], Qu, Kpu, Vpu, N);
    gemm_qkv<<<rowBlocks, 512, 0, stream>>>(h_item, pw[3], pw[4], pw[5], Qi, Kpi, Vpi, N);

    // --- init accumulators ---
    init_buffers<<<(N * HID + 255) / 256, 256, 0, stream>>>(
        m_ui, m_iu, den_ui, den_iu, agg_item, agg_user, N);

    // --- relation user->item (dst = item) and item->user (dst = user) ---
    edge_score<<<(E_ui + 7) / 8, 256, 0, stream>>>(Kpu, Qi, src_ui, dst_ui, mu_ui, s_ui, m_ui, E_ui);
    edge_score<<<(E_iu + 7) / 8, 256, 0, stream>>>(Kpi, Qu, src_iu, dst_iu, mu_iu, s_iu, m_iu, E_iu);
    edge_accum<<<(E_ui + 7) / 8, 256, 0, stream>>>(Vpu, s_ui, m_ui, src_ui, dst_ui, den_ui, agg_item, E_ui);
    edge_accum<<<(E_iu + 7) / 8, 256, 0, stream>>>(Vpi, s_iu, m_iu, src_iu, dst_iu, den_iu, agg_user, E_iu);

    // --- normalize + residual + LayerNorm + ELU ---
    float* out = (float*)d_out;
    out_block<<<(N + 7) / 8, 256, 0, stream>>>(agg_user, den_iu, h_user, ln_g_u, ln_b_u, out, N);
    out_block<<<(N + 7) / 8, 256, 0, stream>>>(agg_item, den_ui, h_item, ln_g_i, ln_b_i, out + (size_t)N * HID, N);
}
